// GroupQueryAttention_84146999263678
// MI455X (gfx1250) — matrix operand order repaired and checked by reading, not compiled, not run on hardware
//
#include <hip/hip_runtime.h>
#include <hip/hip_fp16.h>

typedef __attribute__((ext_vector_type(16))) _Float16 v16h;
typedef __attribute__((ext_vector_type(8)))  _Float16 v8h;
typedef __attribute__((ext_vector_type(8)))  float    v8f;

#define SEQ  4096
#define EMB  2048
#define NH   16
#define NKV  4
#define HD   128

// merge two contiguous 8xf16 LDS chunks into one A-layout fragment
__device__ __forceinline__ v16h make_afrag(const _Float16* lo, const _Float16* hi) {
    v8h a = *(const v8h*)lo;
    v8h b = *(const v8h*)hi;
    return __builtin_shufflevector(a, b, 0, 1, 2, 3, 4, 5, 6, 7,
                                         8, 9, 10, 11, 12, 13, 14, 15);
}

// ---------------------------------------------------------------------------
// C[M,N] = A[M,K] @ B[K,N], fp32 in/out, f16 WMMA accumulate-f32.
// 128x128 block tile, KT=32, 256 threads = 8 waves, wave tile 32x64.
// B staged TRANSPOSED in LDS so B fragments are two contiguous 16B reads.
// ---------------------------------------------------------------------------
__global__ __launch_bounds__(256) void gemm_f16_wmma(
    const float* __restrict__ A, const float* __restrict__ B,
    float* __restrict__ C, int M, int N, int K)
{
    __shared__ _Float16 As[128][40];    // [m][k]  (row = 80B, 16B aligned)
    __shared__ _Float16 BsT[128][40];   // [n][k]  transposed

    const int tid  = threadIdx.x;
    const int lane = tid & 31;
    const int wave = tid >> 5;
    const int half = lane >> 4;
    const int l16  = lane & 15;
    const int wm   = wave & 3;   // 4 row slabs of 32
    const int wn   = wave >> 2;  // 2 col slabs of 64
    const int m0   = blockIdx.y * 128;
    const int n0   = blockIdx.x * 128;

    v8f zero = {};
    v8f acc[2][4];
#pragma unroll
    for (int i = 0; i < 2; ++i)
#pragma unroll
        for (int j = 0; j < 4; ++j) acc[i][j] = zero;

    for (int k0 = 0; k0 < K; k0 += 32) {
        // prefetch next k-tile (global_prefetch_b8)
        if (k0 + 32 < K) {
            __builtin_prefetch(A + (size_t)(m0 + (tid >> 1)) * K + k0 + 32 + (tid & 1) * 16, 0, 0);
            __builtin_prefetch(B + (size_t)(k0 + 32 + (tid >> 3)) * N + n0 + (tid & 7) * 16, 0, 0);
        }
        // stage A tile (128x32) fp32 -> f16, row-major
        for (int i = tid; i < 1024; i += 256) {
            int r = i >> 3, c4 = (i & 7) << 2;
            float4 v = *(const float4*)(A + (size_t)(m0 + r) * K + k0 + c4);
            As[r][c4 + 0] = (_Float16)v.x; As[r][c4 + 1] = (_Float16)v.y;
            As[r][c4 + 2] = (_Float16)v.z; As[r][c4 + 3] = (_Float16)v.w;
        }
        // stage B tile (32x128) fp32 -> f16, TRANSPOSED to [n][k]
        for (int i = tid; i < 1024; i += 256) {
            int r = i >> 5, c4 = (i & 31) << 2;     // r = k row, c4 = col
            float4 v = *(const float4*)(B + (size_t)(k0 + r) * N + n0 + c4);
            BsT[c4 + 0][r] = (_Float16)v.x; BsT[c4 + 1][r] = (_Float16)v.y;
            BsT[c4 + 2][r] = (_Float16)v.z; BsT[c4 + 3][r] = (_Float16)v.w;
        }
        __syncthreads();

        // A fragments: two contiguous 8xf16 chunks per lane
        v16h afrag[2];
#pragma unroll
        for (int mt = 0; mt < 2; ++mt) {
            const _Float16* rowp = &As[wm * 32 + mt * 16 + l16][0];
            afrag[mt] = make_afrag(rowp + half * 8, rowp + 16 + half * 8);
        }
        // B fragments: two contiguous 8xf16 chunks per lane (same K order as A)
        v16h bfrag[4];
#pragma unroll
        for (int nt = 0; nt < 4; ++nt)
            bfrag[nt] = make_afrag(&BsT[wn * 64 + nt * 16 + l16][half * 8],
                                   &BsT[wn * 64 + nt * 16 + l16][16 + half * 8]);

#pragma unroll
        for (int mt = 0; mt < 2; ++mt)
#pragma unroll
            for (int nt = 0; nt < 4; ++nt)
                acc[mt][nt] = __builtin_amdgcn_wmma_f32_16x16x32_f16(
                    false, afrag[mt], false, bfrag[nt], (short)0, acc[mt][nt],
                    false, false);
        __syncthreads();
    }

    // C/D layout: VGPR r holds rows r (lanes 0-15) and r+8 (lanes 16-31)
#pragma unroll
    for (int mt = 0; mt < 2; ++mt)
#pragma unroll
        for (int nt = 0; nt < 4; ++nt)
#pragma unroll
            for (int r = 0; r < 8; ++r) {
                int row = m0 + wm * 32 + mt * 16 + r + half * 8;
                int col = n0 + wn * 64 + nt * 16 + l16;
                C[(size_t)row * N + col] = acc[mt][nt][r];
            }
}

// ---------------------------------------------------------------------------
// RMSNorm(HEAD_DIM=128) + RoPE. One wave per (token, head) vector.
// in: [SEQ][nh*128] row-major   out: [nh][SEQ][128] head-major
// ---------------------------------------------------------------------------
__global__ __launch_bounds__(128) void rms_rope_kernel(
    const float* __restrict__ in, const float* __restrict__ nw,
    const int* __restrict__ pos_ids, float* __restrict__ out, int nh)
{
    const int wave = threadIdx.x >> 5;
    const int lane = threadIdx.x & 31;
    const int gw   = blockIdx.x * 4 + wave;
    const int h    = gw % nh;
    const int s    = gw / nh;
    const int d0   = lane * 4;

    float4 v = *(const float4*)(in + (size_t)s * (nh * HD) + h * HD + d0);
    float ss = v.x * v.x + v.y * v.y + v.z * v.z + v.w * v.w;
#pragma unroll
    for (int m = 1; m < 32; m <<= 1) ss += __shfl_xor(ss, m, 32);
    const float r = rsqrtf(ss * (1.0f / HD) + 1e-6f);
    float4 w4 = *(const float4*)(nw + d0);
    float mv[4] = { v.x * r * w4.x, v.y * r * w4.y, v.z * r * w4.z, v.w * r * w4.w };

    // exchange with the lane holding d^64 (lane ^ 16)
    float ov[4];
#pragma unroll
    for (int c = 0; c < 4; ++c) ov[c] = __shfl_xor(mv[c], 16, 32);

    const float pos = (float)pos_ids[s];
    float res[4];
#pragma unroll
    for (int c = 0; c < 4; ++c) {
        int d = d0 + c;
        int j = d & 63;                       // freq index (cos[j]==cos[j+64])
        float freq = __expf(-(float)j * (9.210340371976184f / 64.0f)); // 10000^(-j/64)
        float ang = pos * freq;
        float cv = __cosf(ang), sv = __sinf(ang);
        res[c] = (d < 64) ? (mv[c] * cv - ov[c] * sv)
                          : (mv[c] * cv + ov[c] * sv);
    }
    float4 o4 = { res[0], res[1], res[2], res[3] };
    *(float4*)(out + ((size_t)h * SEQ + s) * HD + d0) = o4;
}

// V: [SEQ][NKV*128] -> [NKV][SEQ][128]
__global__ __launch_bounds__(256) void repack_v_kernel(
    const float* __restrict__ in, float* __restrict__ out)
{
    int i = blockIdx.x * 256 + threadIdx.x;      // NKV*SEQ*HD total
    int d  = i & (HD - 1);
    int t  = i >> 7;
    int s  = t & (SEQ - 1);
    int kv = t >> 12;
    out[i] = in[(size_t)s * (NKV * HD) + kv * HD + d];
}

// ---------------------------------------------------------------------------
// Flash attention. Grid (SEQ/64, NH), 128 threads = 4 waves, 16 q-rows/wave.
// Q: [NH][SEQ][128] (pre-scaled by 1/sqrt(128) here)  K,V: [NKV][SEQ][128]
// V staged TRANSPOSED so PV B-fragments are two contiguous 16B reads.
// ctx out: [SEQ][NH*128]
// ---------------------------------------------------------------------------
__global__ __launch_bounds__(128) void attn_kernel(
    const float* __restrict__ Qf, const float* __restrict__ Kc,
    const float* __restrict__ Vc, float* __restrict__ ctx)
{
    __shared__ _Float16 Qs[64][136];    // [q][d]
    __shared__ _Float16 Ks[32][136];    // [key][d]   (K^T fragments contiguous)
    __shared__ _Float16 VsT[128][40];   // [d][key]   transposed
    __shared__ _Float16 Ps[4][16][40];  // per-wave P bounce

    const int h    = blockIdx.y;
    const int kvh  = h >> 2;               // 16 heads / 4 kv heads
    const int q0   = blockIdx.x * 64;
    const int tid  = threadIdx.x;
    const int lane = tid & 31;
    const int wave = tid >> 5;
    const int half = lane >> 4;
    const int l16  = lane & 15;
    const float scale = 0.08838834764831845f;   // 1/sqrt(128)

    // stage Q tile 64x128 (scale folded in)
    const float* Qh = Qf + ((size_t)h * SEQ + q0) * HD;
    for (int i = tid; i < 2048; i += 128) {
        int r = i >> 5, c4 = (i & 31) << 2;
        float4 v = *(const float4*)(Qh + (size_t)r * HD + c4);
        Qs[r][c4 + 0] = (_Float16)(v.x * scale);
        Qs[r][c4 + 1] = (_Float16)(v.y * scale);
        Qs[r][c4 + 2] = (_Float16)(v.z * scale);
        Qs[r][c4 + 3] = (_Float16)(v.w * scale);
    }
    __syncthreads();

    // cache Q A-fragments in registers: 4 K-steps of 32
    v16h qfrag[4];
    {
        const _Float16* rowp = &Qs[wave * 16 + l16][0];
#pragma unroll
        for (int ks = 0; ks < 4; ++ks)
            qfrag[ks] = make_afrag(rowp + ks * 32 + half * 8,
                                   rowp + ks * 32 + 16 + half * 8);
    }

    v8f zero = {};
    v8f o[8];
#pragma unroll
    for (int j = 0; j < 8; ++j) o[j] = zero;
    float Mrow[8], Lrow[8];
#pragma unroll
    for (int r = 0; r < 8; ++r) { Mrow[r] = -1e30f; Lrow[r] = 0.0f; }

    const int nkb = (q0 >> 5) + 2;   // key blocks of 32 covering keys <= q0+63
    for (int kb = 0; kb < nkb; ++kb) {
        const int kbase = kb * 32;
        __syncthreads();             // previous iter done with Ks/VsT
        if (kb + 1 < nkb)            // prefetch next key block
            __builtin_prefetch(Kc + ((size_t)kvh * SEQ + kbase + 32 + (tid >> 2)) * HD
                                   + (tid & 3) * 32, 0, 0);
        for (int i = tid; i < 1024; i += 128) {
            int r = i >> 5, c4 = (i & 31) << 2;   // r = key, c4 = d
            float4 kv4 = *(const float4*)(Kc + ((size_t)kvh * SEQ + kbase + r) * HD + c4);
            Ks[r][c4 + 0] = (_Float16)kv4.x; Ks[r][c4 + 1] = (_Float16)kv4.y;
            Ks[r][c4 + 2] = (_Float16)kv4.z; Ks[r][c4 + 3] = (_Float16)kv4.w;
            float4 vv4 = *(const float4*)(Vc + ((size_t)kvh * SEQ + kbase + r) * HD + c4);
            VsT[c4 + 0][r] = (_Float16)vv4.x; VsT[c4 + 1][r] = (_Float16)vv4.y;
            VsT[c4 + 2][r] = (_Float16)vv4.z; VsT[c4 + 3][r] = (_Float16)vv4.w;
        }
        __syncthreads();

        // S = Q @ K^T : B[k][n] = K[n][k] -> two 8xf16 chunks in Ks rows
        v8f sa[2];
        sa[0] = zero; sa[1] = zero;
#pragma unroll
        for (int nt = 0; nt < 2; ++nt) {
            const _Float16* keyp = &Ks[nt * 16 + l16][0];
#pragma unroll
            for (int ks = 0; ks < 4; ++ks) {
                v16h b = make_afrag(keyp + ks * 32 + half * 8, keyp + ks * 32 + 16 + half * 8);
                sa[nt] = __builtin_amdgcn_wmma_f32_16x16x32_f16(
                    false, qfrag[ks], false, b, (short)0, sa[nt], false, false);
            }
        }

        // causal mask (same -1e9 as reference's additive mask)
#pragma unroll
        for (int nt = 0; nt < 2; ++nt)
#pragma unroll
            for (int r = 0; r < 8; ++r) {
                int key = kbase + nt * 16 + l16;
                int q   = q0 + wave * 16 + r + half * 8;
                float s = sa[nt][r];
                sa[nt][r] = (key > q) ? -1e9f : s;
            }

        // online softmax; rows live in 16-lane halves -> masks 1,2,4,8 stay inside
        float alpha[8];
#pragma unroll
        for (int r = 0; r < 8; ++r) {
            float mx = fmaxf(sa[0][r], sa[1][r]);
#pragma unroll
            for (int m = 1; m < 16; m <<= 1) mx = fmaxf(mx, __shfl_xor(mx, m, 32));
            float mnew = fmaxf(Mrow[r], mx);
            float a  = __expf(Mrow[r] - mnew);
            float p0 = __expf(sa[0][r] - mnew);
            float p1 = __expf(sa[1][r] - mnew);
            float ps = p0 + p1;
#pragma unroll
            for (int m = 1; m < 16; m <<= 1) ps += __shfl_xor(ps, m, 32);
            Lrow[r] = Lrow[r] * a + ps;
            Mrow[r] = mnew;
            alpha[r] = a;
            sa[0][r] = p0;
            sa[1][r] = p1;
        }
#pragma unroll
        for (int j = 0; j < 8; ++j)
#pragma unroll
            for (int r = 0; r < 8; ++r) o[j][r] *= alpha[r];

        // P: C-layout -> A-layout via per-wave LDS bounce
#pragma unroll
        for (int nt = 0; nt < 2; ++nt)
#pragma unroll
            for (int r = 0; r < 8; ++r)
                Ps[wave][r + half * 8][nt * 16 + l16] = (_Float16)sa[nt][r];
        __syncthreads();

        v16h pfrag;
        {
            const _Float16* rowp = &Ps[wave][l16][0];
            pfrag = make_afrag(rowp + half * 8, rowp + 16 + half * 8);
        }
        // O += P @ V : B[k][n] = V[key][d] -> two 8xf16 chunks in VsT rows
#pragma unroll
        for (int j = 0; j < 8; ++j) {
            v16h b = make_afrag(&VsT[j * 16 + l16][half * 8], &VsT[j * 16 + l16][16 + half * 8]);
            o[j] = __builtin_amdgcn_wmma_f32_16x16x32_f16(
                false, pfrag, false, b, (short)0, o[j], false, false);
        }
    }

    // normalize and scatter ctx[s][h*128 + d]
#pragma unroll
    for (int r = 0; r < 8; ++r) {
        float inv = 1.0f / Lrow[r];
        int q = q0 + wave * 16 + r + half * 8;
#pragma unroll
        for (int j = 0; j < 8; ++j)
            ctx[(size_t)q * (NH * HD) + h * HD + j * 16 + l16] = o[j][r] * inv;
    }
}

// ---------------------------------------------------------------------------
extern "C" void kernel_launch(void* const* d_in, const int* in_sizes, int n_in,
                              void* d_out, int out_size, void* d_ws, size_t ws_size,
                              hipStream_t stream)
{
    const float* x   = (const float*)d_in[0];
    const int*   pos = (const int*)d_in[1];
    // d_in[2] = attn_mask (causal -1e9 upper triangle): computed analytically
    const float* Wq  = (const float*)d_in[3];
    const float* Wk  = (const float*)d_in[4];
    const float* Wv  = (const float*)d_in[5];
    const float* Wo  = (const float*)d_in[6];
    const float* qnw = (const float*)d_in[7];
    const float* knw = (const float*)d_in[8];

    float* out  = (float*)d_out;                       // [SEQ][EMB]
    float* Kout = out + (size_t)SEQ * EMB;             // [NKV][SEQ][HD]
    float* Vout = Kout + (size_t)NKV * SEQ * HD;       // [NKV][SEQ][HD]

    float* ws  = (float*)d_ws;
    float* Qp  = ws;                                   // [SEQ][NH*HD]  32MB
    float* Kp  = Qp + (size_t)SEQ * NH * HD;           // [SEQ][NKV*HD]  8MB
    float* Vp  = Kp + (size_t)SEQ * NKV * HD;          // [SEQ][NKV*HD]  8MB
    float* Qf  = Vp + (size_t)SEQ * NKV * HD;          // [NH][SEQ][HD] 32MB
    float* ctx = Qp;                                   // reuse Qp after Qf built

    // 1-3) projections
    gemm_f16_wmma<<<dim3(EMB / 128, SEQ / 128), 256, 0, stream>>>(x, Wq, Qp, SEQ, NH * HD, EMB);
    gemm_f16_wmma<<<dim3((NKV * HD) / 128, SEQ / 128), 256, 0, stream>>>(x, Wk, Kp, SEQ, NKV * HD, EMB);
    gemm_f16_wmma<<<dim3((NKV * HD) / 128, SEQ / 128), 256, 0, stream>>>(x, Wv, Vp, SEQ, NKV * HD, EMB);

    // 4-6) RMSNorm + RoPE (Q -> scratch head-major, K -> KV-cache output), V repack
    rms_rope_kernel<<<(SEQ * NH) / 4, 128, 0, stream>>>(Qp, qnw, pos, Qf, NH);
    rms_rope_kernel<<<(SEQ * NKV) / 4, 128, 0, stream>>>(Kp, knw, pos, Kout, NKV);
    repack_v_kernel<<<(NKV * SEQ * HD) / 256, 256, 0, stream>>>(Vp, Vout);

    // 7) flash attention -> ctx [SEQ][NH*HD]
    attn_kernel<<<dim3(SEQ / 64, NH), 128, 0, stream>>>(Qf, Kout, Vout, ctx);

    // 8) output projection
    gemm_f16_wmma<<<dim3(EMB / 128, SEQ / 128), 256, 0, stream>>>(ctx, Wo, out, SEQ, EMB, EMB);
}
